// KANCouplingNet_40656160424367
// MI455X (gfx1250) — compile-verified, matrix operand order checked by reading, not run on hardware
//
#include <hip/hip_runtime.h>

// ---------- types ----------
typedef __attribute__((ext_vector_type(16))) __bf16        v16bf;
typedef __attribute__((ext_vector_type(8)))  float         v8f;
typedef __attribute__((ext_vector_type(4)))  unsigned int  u32x4;   // trivial 16B vector (safe in unions)

union Frag32B { u32x4 q[2]; v16bf v; };                  // one WMMA 16-bf16 operand fragment per lane
union Pack16B { unsigned short s[8]; u32x4 q; };         // 8 bf16 packed for one b128 LDS store

__device__ __forceinline__ unsigned short f2bf(float f) {
  unsigned int u = __float_as_uint(f);
  u += 0x7FFFu + ((u >> 16) & 1u);                       // round-to-nearest-even
  return (unsigned short)(u >> 16);
}

// ---------- weight pre-pack: fold sb / sp*coef into bf16, col-major [Npad][Kpad] ----------
// K layout per input feature i: k = 8*i + g  -> sp[i,o]*coef[i,o,g]   (g = 0..7)
//                               k = 8*n_in+i -> sb[i,o]               (silu column)
__global__ void kan_pack_weights(const float* __restrict__ coef,
                                 const float* __restrict__ sb,
                                 const float* __restrict__ sp,
                                 unsigned short* __restrict__ wout,
                                 int n_in, int n_out, int Npad, int Kpad) {
  const int total = Npad * Kpad;
  for (int idx = blockIdx.x * blockDim.x + threadIdx.x; idx < total;
       idx += gridDim.x * blockDim.x) {
    const int o = idx / Kpad;
    const int k = idx - o * Kpad;
    float v = 0.0f;
    if (o < n_out) {
      if (k < 8 * n_in) {
        const int i = k >> 3, g = k & 7;
        v = sp[i * n_out + o] * coef[(i * n_out + o) * 8 + g];
      } else if (k < 9 * n_in) {
        const int i = k - 8 * n_in;
        v = sb[i * n_out + o];
      }
    }
    wout[idx] = f2bf(v);
  }
}

// ---------- one fused KAN layer: expand (silu + uniform cubic B-spline) -> bf16 WMMA GEMM ----------
template<int N_IN, int KPAD, int NTILES>
__device__ __forceinline__ void kan_layer_step(const unsigned short* __restrict__ gW,
                                               unsigned short* sA,
                                               unsigned short* sW,
                                               float* sH, int tid) {
  // stage packed bf16 weights (col-major [NTILES*16][KPAD]) into LDS, 16B vectors
  {
    const int NQ = (NTILES * 16 * KPAD) / 8;
    u32x4* d = (u32x4*)sW;
    const u32x4* s = (const u32x4*)gW;
    for (int i = tid; i < NQ; i += 256) d[i] = s[i];
  }

  // expand 128 rows x N_IN features -> A tile [128][KPAD] bf16
  for (int t = tid; t < 128 * N_IN; t += 256) {
    const int row = t / N_IN;
    const int i   = t - row * N_IN;
    const float xv = sH[row * 64 + i];
    const float sl = xv / (1.0f + __expf(-xv));          // silu
    // uniform grid: h = 2/5 = 0.4, cell c = floor((x+1)/h), u = frac
    const float tt = (xv + 1.0f) * 2.5f;
    const float fc = floorf(tt);
    const int   c  = (int)fc;
    const float u  = tt - fc;
    const float um = 1.0f - u;
    const float u2 = u * u, u3 = u2 * u;
    float w[4];
    w[0] = um * um * um * (1.0f / 6.0f);
    w[1] = (3.0f * u3 - 6.0f * u2 + 4.0f) * (1.0f / 6.0f);
    w[2] = (-3.0f * u3 + 3.0f * u2 + 3.0f * u + 1.0f) * (1.0f / 6.0f);
    w[3] = u3 * (1.0f / 6.0f);
    Pack16B pk;
#pragma unroll
    for (int g = 0; g < 8; ++g) {                        // basis g nonzero only for g-c in [0,4)
      const int d = g - c;
      pk.s[g] = f2bf((d >= 0 && d < 4) ? w[d] : 0.0f);
    }
    ((u32x4*)sA)[(row * KPAD + 8 * i) >> 3] = pk.q;      // 8 basis bf16 in one b128
    sA[row * KPAD + 8 * N_IN + i] = f2bf(sl);            // silu column
  }
  if (KPAD > 9 * N_IN) {                                 // zero K padding
    const int PAD = KPAD - 9 * N_IN;
    for (int t = tid; t < 128 * PAD; t += 256) {
      const int row = t / PAD;
      sA[row * KPAD + 9 * N_IN + (t - row * PAD)] = 0;
    }
  }
  __syncthreads();

  // WMMA GEMM: wave wv owns rows [16*wv, 16*wv+16), all NTILES column tiles.
  // Software-pipelined: double-buffered fragments so ds_loads for step k+32
  // are in flight while the WMMAs for step k execute (avoids s_wait_dscnt 0).
  const int wv   = tid >> 5;
  const int lane = tid & 31;
  const int half = lane >> 4;                            // K-half select per ISA A/B 16-bit layout
  const int lrow = lane & 15;
  v8f acc[NTILES];
#pragma unroll
  for (int nt = 0; nt < NTILES; ++nt) acc[nt] = {};

  const u32x4* A4 = (const u32x4*)sA;
  const u32x4* W4 = (const u32x4*)sW;
  const int arow = wv * 16 + lrow;

  auto loadA = [&](Frag32B& f, int kk) {
    const int ab = (arow * KPAD + kk + 8 * half) >> 3;   // elems 0..7 = K[kk+8h..], 8..15 = K[kk+16+8h..]
    f.q[0] = A4[ab];
    f.q[1] = A4[ab + 2];
  };
  auto loadB = [&](Frag32B* f, int kk) {
#pragma unroll
    for (int nt = 0; nt < NTILES; ++nt) {
      const int bb = ((nt * 16 + lrow) * KPAD + kk + 8 * half) >> 3;
      f[nt].q[0] = W4[bb];
      f[nt].q[1] = W4[bb + 2];
    }
  };
  auto domma = [&](Frag32B& fa, Frag32B* fb) {
#pragma unroll
    for (int nt = 0; nt < NTILES; ++nt)
      acc[nt] = __builtin_amdgcn_wmma_f32_16x16x32_bf16(
          false, fa.v, false, fb[nt].v, (short)0, acc[nt], false, false);
  };

  Frag32B fa0, fa1, fb0[NTILES], fb1[NTILES];
  loadA(fa0, 0);
  loadB(fb0, 0);
  static_assert((KPAD / 32) % 2 == 0, "pipeline assumes even step count");
  for (int kk = 0; kk < KPAD; kk += 64) {
    // prefetch step kk+32 while computing step kk
    loadA(fa1, kk + 32);
    loadB(fb1, kk + 32);
    domma(fa0, fb0);
    // prefetch step kk+64 while computing step kk+32
    if (kk + 64 < KPAD) {
      loadA(fa0, kk + 64);
      loadB(fb0, kk + 64);
    }
    domma(fa1, fb1);
  }

  // C/D layout: VGPR r, lanes 0-15 -> M=r, lanes 16-31 -> M=8+r; N = lane&15
#pragma unroll
  for (int nt = 0; nt < NTILES; ++nt)
#pragma unroll
    for (int r = 0; r < 8; ++r)
      sH[(wv * 16 + r + (half << 3)) * 64 + nt * 16 + lrow] = acc[nt][r];
  __syncthreads();
}

// ---------- fused 3-layer KAN over 128 rows per block ----------
__global__ __launch_bounds__(256, 1)
void kan_fused(const float* __restrict__ x,
               const unsigned short* __restrict__ wpack,
               float* __restrict__ out) {
  extern __shared__ char smem[];
  unsigned short* sA = (unsigned short*)smem;                    // 128*576 bf16 = 147456 B
  unsigned short* sW = (unsigned short*)(smem + 128 * 576 * 2);  //  64*576 bf16 =  73728 B
  float*          sH = (float*)(smem + 128 * 576 * 2 + 64 * 576 * 2); // 128*64 f32 = 32768 B

  const int tid  = threadIdx.x;
  const int row0 = blockIdx.x * 128;     // rows = pixels in NHWC order
  const int n    = row0 >> 12;           // H*W = 4096; 128 | 4096 -> one image per block
  const int p0   = row0 & 4095;

  // coalesced NCHW gather of 12 input channels for 128 consecutive pixels
  for (int t = tid; t < 12 * 128; t += 256) {
    const int c = t >> 7, tr = t & 127;
    sH[tr * 64 + c] = x[((n * 12 + c) << 12) + p0 + tr];
  }
  __syncthreads();

  kan_layer_step<12, 128, 4>(wpack,                 sA, sW, sH, tid);  // 12 -> 64
  kan_layer_step<64, 576, 4>(wpack + 8192,          sA, sW, sH, tid);  // 64 -> 64
  kan_layer_step<64, 576, 2>(wpack + 8192 + 36864,  sA, sW, sH, tid);  // 64 -> 24 (Npad 32)

  // coalesced NCHW scatter of 24 output channels
  for (int t = tid; t < 24 * 128; t += 256) {
    const int o = t >> 7, tr = t & 127;
    out[((n * 24 + o) << 12) + p0 + tr] = sH[tr * 64 + o];
  }
}

extern "C" void kernel_launch(void* const* d_in, const int* in_sizes, int n_in,
                              void* d_out, int out_size, void* d_ws, size_t ws_size,
                              hipStream_t stream) {
  const float* x     = (const float*)d_in[0];
  const float* coef0 = (const float*)d_in[2];
  const float* sb0   = (const float*)d_in[3];
  const float* sp0   = (const float*)d_in[4];
  const float* coef1 = (const float*)d_in[6];
  const float* sb1   = (const float*)d_in[7];
  const float* sp1   = (const float*)d_in[8];
  const float* coef2 = (const float*)d_in[10];
  const float* sb2   = (const float*)d_in[11];
  const float* sp2   = (const float*)d_in[12];

  unsigned short* wpack = (unsigned short*)d_ws;   // 63488 bf16 = 124 KB total

  kan_pack_weights<<<32,  256, 0, stream>>>(coef0, sb0, sp0, wpack,         12, 64, 64, 128);
  kan_pack_weights<<<144, 256, 0, stream>>>(coef1, sb1, sp1, wpack + 8192,  64, 64, 64, 576);
  kan_pack_weights<<<72,  256, 0, stream>>>(coef2, sb2, sp2, wpack + 45056, 64, 24, 32, 576);

  const size_t smem = 128 * 576 * 2 + 64 * 576 * 2 + 128 * 64 * 4;  // 253952 B < 320 KB LDS
  (void)hipFuncSetAttribute((const void*)kan_fused,
                            hipFuncAttributeMaxDynamicSharedMemorySize, (int)smem);
  kan_fused<<<1024, 256, smem, stream>>>(x, wpack, (float*)d_out);
}
